// ConvNetLayer_61074434949126
// MI455X (gfx1250) — matrix operand order repaired and checked by reading, not compiled, not run on hardware
//
#include <hip/hip_runtime.h>
#include <hip/hip_bf16.h>

// ---------------------------------------------------------------------------
// Problem constants (match reference)
// ---------------------------------------------------------------------------
#define MULC      32
#define N_RAD     8
#define FC_H      64
#define INV_SQRT8   0.35355339059327373f   // 1/sqrt(8)
#define INV_8       0.125f                 // 1/sqrt(64)
#define INV_SQRT32  0.17677669529663687f   // 1/sqrt(32)
#define INV3C       0.5773502691896258f    // 1/sqrt(3)
#define INV_AVG     0.0625f                // 1/16
#define NORM_SC     0.08838834764831845f   // 1/sqrt(128)

#define W16_TOTAL   (64 * 32 + 64 * 64 + 128 * 64)   // 14336 f16 elements (28 KB)
#define WF2_OFF     (64 * 32)
#define WF3_OFF     (64 * 32 + 64 * 64)

typedef __attribute__((ext_vector_type(16))) _Float16 v16h;
typedef __attribute__((ext_vector_type(8)))  _Float16 h8;
typedef __attribute__((ext_vector_type(8)))  float    v8f;
typedef __attribute__((ext_vector_type(4)))  unsigned int u32x4;
typedef __attribute__((ext_vector_type(8)))  int      i32x8;
typedef __attribute__((ext_vector_type(4)))  int      i32x4;

// Fast SiLU: x * v_rcp_f32(1 + e^-x)  (single HW reciprocal, no div ladder)
__device__ __forceinline__ float silu_f(float x) {
    return x * __builtin_amdgcn_rcpf(1.0f + __expf(-x));
}

// A-fragment (16x32 f16, row-major source, row stride ldk elements).
// Lane L: row = L&15, half = L>>4.
// elements 0..15 <- K = half*16 + 0..15 (contiguous)
__device__ __forceinline__ v16h load_fragA(const _Float16* __restrict__ base,
                                           int row, int ldk, int lane) {
    const _Float16* p = base + row * ldk;
    int h8off = (lane >> 4) << 4;
    h8 lo = *(const h8*)(p + h8off);
    h8 hi = *(const h8*)(p + 8 + h8off);
    v16h f;
#pragma unroll
    for (int i = 0; i < 8; ++i) { f[i] = lo[i]; f[i + 8] = hi[i]; }
    return f;
}

// B-fragment (32x16 f16). Weights stored transposed: [out_n][in_k].
// Lane L: col N = L&15, elements 0..15 <- K = (L>>4)*16 + 0..15 (contiguous).
__device__ __forceinline__ v16h load_fragB(const _Float16* __restrict__ baseT,
                                           int rowN, int ldk, int lane) {
    const _Float16* p = baseT + rowN * ldk + ((lane >> 4) << 4);
    return *(const v16h*)p;
}

__device__ __forceinline__ v8f wmma16(v16h a, v16h b, v8f c) {
    return __builtin_amdgcn_wmma_f32_16x16x32_f16(
        false, a, false, b, (short)0, c, false, false);
}

// ---------------------------------------------------------------------------
// Kernel W: transpose + cast radial-MLP weights to f16 (K padded to 32 for L1)
// One contiguous blob: [wf1t | wf2t | wf3t]
// ---------------------------------------------------------------------------
__global__ __launch_bounds__(256)
void prep_weights(const float* __restrict__ Wf1, const float* __restrict__ Wf2,
                  const float* __restrict__ Wf3, _Float16* __restrict__ wall) {
    int t = threadIdx.x;
    for (int i = t; i < 64 * 32; i += 256) {           // wf1t[n][k], k<8 valid
        int n = i >> 5, k = i & 31;
        wall[i] = (_Float16)((k < N_RAD) ? Wf1[k * 64 + n] : 0.0f);
    }
    for (int i = t; i < 64 * 64; i += 256) {           // wf2t[n][k]
        int n = i >> 6, k = i & 63;
        wall[WF2_OFF + i] = (_Float16)Wf2[k * 64 + n];
    }
    for (int i = t; i < 128 * 64; i += 256) {          // wf3t[n][k]
        int n = i >> 6, k = i & 63;
        wall[WF3_OFF + i] = (_Float16)Wf3[k * 128 + n];
    }
}

// ---------------------------------------------------------------------------
// Kernel A: node pre-linears  xs = xs0@W1s/sqrt(32), xv = xv0 x W1v/sqrt(32)
// xs layout [n][u]; xv layout [n][i][u]  (u fastest -> coalesced gathers)
// ---------------------------------------------------------------------------
__global__ __launch_bounds__(256)
void node_pre(const float* __restrict__ nf, const float* __restrict__ W1s,
              const float* __restrict__ W1v,
              float* __restrict__ xs, float* __restrict__ xv) {
    int gid = blockIdx.x * 256 + threadIdx.x;
    int n = gid >> 5, u = gid & 31;
    const float* row = nf + (size_t)n * 128;
    float s = 0.f, vx = 0.f, vy = 0.f, vz = 0.f;
#pragma unroll 4
    for (int k = 0; k < MULC; ++k) {
        float ws = W1s[k * 32 + u];
        float wv = W1v[k * 32 + u];
        s  += row[k] * ws;
        vx += row[32 + k * 3 + 0] * wv;
        vy += row[32 + k * 3 + 1] * wv;
        vz += row[32 + k * 3 + 2] * wv;
    }
    xs[(size_t)n * 32 + u] = s * INV_SQRT32;
    xv[((size_t)n * 3 + 0) * 32 + u] = vx * INV_SQRT32;
    xv[((size_t)n * 3 + 1) * 32 + u] = vy * INV_SQRT32;
    xv[((size_t)n * 3 + 2) * 32 + u] = vz * INV_SQRT32;
}

// ---------------------------------------------------------------------------
// Kernel E: fused edge MLP (WMMA) + message build + atomic scatter
// One wave = 16 edges.  4 waves / block = 64 edges / block.
// Weights staged into LDS once per block via the Tensor Data Mover.
// ---------------------------------------------------------------------------
#define WPB 4
__global__ __launch_bounds__(WPB * 32)
void edge_kernel(const float* __restrict__ ef, const float* __restrict__ ea,
                 const int* __restrict__ eidx,
                 const float* __restrict__ xs, const float* __restrict__ xv,
                 const _Float16* __restrict__ wall,
                 float* __restrict__ aggs, float* __restrict__ aggv, int E_) {
    __shared__ _Float16 wlds[W16_TOTAL];     // 28 KB: all three weight matrices
    __shared__ _Float16 hbuf[WPB][16][64];   // per-wave hidden tile (f16)
    __shared__ _Float16 wbuf[WPB][16][128];  // per-wave final MLP outputs (f16)
    __shared__ int      sidx[WPB][16];       // senders
    __shared__ int      ridx[WPB][16];       // receivers
    __shared__ float4   sattr[WPB][16];      // edge spherical harmonics

    const int lane = threadIdx.x & 31;
    const int wv   = threadIdx.x >> 5;
    const int e0   = (blockIdx.x * WPB + wv) * 16;
    const int rowm = lane & 15;
    const int half = lane >> 4;

    // ---- Stage weight blob (global f16 -> LDS) with the TDM, wave 0 only.
    if (threadIdx.x < 32) {
#if __has_builtin(__builtin_amdgcn_tensor_load_to_lds)
        const unsigned ldsa = (unsigned)(size_t)(void*)&wlds[0];
        const unsigned long long ga = (unsigned long long)(size_t)wall;
        const unsigned nel = (unsigned)W16_TOTAL;       // 14336 elements of 2 B
        u32x4 g0;
        g0[0] = 1u;                                     // count=1, user mode
        g0[1] = ldsa;                                   // lds_addr (bytes)
        g0[2] = (unsigned)(ga & 0xFFFFFFFFu);           // global_addr[31:0]
        g0[3] = (unsigned)((ga >> 32) & 0x01FFFFFFu)    // global_addr[56:32]
              | (2u << 30);                             // type = 2 (image)
        i32x8 g1;
        g1[0] = (int)(1u << 16);                        // data_size=1 (2 bytes)
        g1[1] = (int)((nel & 0xFFFFu) << 16);           // tensor_dim0[15:0]
        g1[2] = (int)((nel >> 16) | (1u << 16));        // tensor_dim0[31:16] | tensor_dim1=1
        g1[3] = (int)(nel << 16);                       // tile_dim0 = nel
        g1[4] = 1;                                      // tile_dim1=1, tile_dim2=0
        g1[5] = (int)nel;                               // tensor_dim0_stride lo
        g1[6] = 0;
        g1[7] = 0;
        i32x4 gz = {0, 0, 0, 0};
#if defined(__clang_major__) && (__clang_major__ >= 23)
        i32x8 gz8 = {0, 0, 0, 0, 0, 0, 0, 0};
        __builtin_amdgcn_tensor_load_to_lds(g0, g1, gz, gz, gz8, 0);
#else
        __builtin_amdgcn_tensor_load_to_lds(g0, g1, gz, gz, 0);
#endif
        __builtin_amdgcn_s_wait_tensorcnt(0);
#else
        for (int i = lane; i < W16_TOTAL; i += 32) wlds[i] = wall[i];
#endif
    }
    // ---- Per-wave edge metadata into LDS (lanes 0..15), broadcast later.
    if (lane < 16) {
        const int eg = e0 + lane;
        sidx[wv][lane]  = eidx[eg];
        ridx[wv][lane]  = eidx[E_ + eg];
        sattr[wv][lane] = *(const float4*)(ea + (size_t)eg * 4);
    }
    __builtin_prefetch(ef + (size_t)e0 * N_RAD, 0, 1);
    __syncthreads();

    const _Float16* wf1t = &wlds[0];
    const _Float16* wf2t = &wlds[WF2_OFF];
    const _Float16* wf3t = &wlds[WF3_OFF];

    // ---- Layer 1: (16x8 padded to 16x32) @ (32x64), 1/sqrt(8) folded into A
    v16h a1 = {};
    if (half == 0) {
        const float* fe = ef + (size_t)(e0 + rowm) * N_RAD;
#pragma unroll
        for (int k = 0; k < N_RAD; ++k)
            a1[k] = (_Float16)(fe[k] * INV_SQRT8);
    }
    v8f acc[8];
    const v8f zacc = {};
#pragma unroll
    for (int t = 0; t < 4; ++t) {
        v16h b = load_fragB(wf1t, t * 16 + rowm, 32, lane);
        acc[t] = wmma16(a1, b, zacc);
    }
#pragma unroll
    for (int t = 0; t < 4; ++t)
#pragma unroll
        for (int j = 0; j < 8; ++j)
            hbuf[wv][j + 8 * half][t * 16 + rowm] = (_Float16)silu_f(acc[t][j]);

    // ---- Layer 2: (16x64) @ (64x64), scale 1/8 on output
    v16h a20 = load_fragA(&hbuf[wv][0][0], rowm, 64, lane);
    v16h a21 = load_fragA(&hbuf[wv][0][0] + 32, rowm, 64, lane);
#pragma unroll
    for (int t = 0; t < 4; ++t) {
        v16h b0 = load_fragB(wf2t,      t * 16 + rowm, 64, lane);
        v16h b1 = load_fragB(wf2t + 32, t * 16 + rowm, 64, lane);
        v8f c = wmma16(a20, b0, zacc);
        acc[t] = wmma16(a21, b1, c);
    }
#pragma unroll
    for (int t = 0; t < 4; ++t)
#pragma unroll
        for (int j = 0; j < 8; ++j)
            hbuf[wv][j + 8 * half][t * 16 + rowm] =
                (_Float16)silu_f(acc[t][j] * INV_8);

    // ---- Layer 3: (16x64) @ (64x128), scale 1/8 on output
    v16h a30 = load_fragA(&hbuf[wv][0][0], rowm, 64, lane);
    v16h a31 = load_fragA(&hbuf[wv][0][0] + 32, rowm, 64, lane);
#pragma unroll
    for (int t = 0; t < 8; ++t) {
        v16h b0 = load_fragB(wf3t,      t * 16 + rowm, 64, lane);
        v16h b1 = load_fragB(wf3t + 32, t * 16 + rowm, 64, lane);
        v8f c = wmma16(a30, b0, zacc);
        acc[t] = wmma16(a31, b1, c);
    }
#pragma unroll
    for (int t = 0; t < 8; ++t)
#pragma unroll
        for (int j = 0; j < 8; ++j)
            wbuf[wv][j + 8 * half][t * 16 + rowm] =
                (_Float16)(acc[t][j] * INV_8);

    // ---- Message phase: lane = channel u, loop over the wave's 16 edges.
    const int u = lane;
#pragma unroll 2
    for (int e = 0; e < 16; ++e) {
        const int s = sidx[wv][e];
        const int r = ridx[wv][e];
        const float4 sh = sattr[wv][e];
        const float sh0 = sh.x, s1x = sh.y, s1y = sh.z, s1z = sh.w;
        const float w0 = (float)wbuf[wv][e][u];
        const float w1 = (float)wbuf[wv][e][32 + u];
        const float w2 = (float)wbuf[wv][e][64 + u];
        const float w3 = (float)wbuf[wv][e][96 + u];
        const float xse = xs[(size_t)s * 32 + u];
        const float xvx = xv[((size_t)s * 3 + 0) * 32 + u];
        const float xvy = xv[((size_t)s * 3 + 1) * 32 + u];
        const float xvz = xv[((size_t)s * 3 + 2) * 32 + u];

        const float ms0  = w0 * xse * sh0 * INV_AVG;
        const float dotv = xvx * s1x + xvy * s1y + xvz * s1z;
        const float ms1  = w3 * dotv * INV3C * INV_AVG;
        const float w1x  = w1 * xse * INV_AVG;
        const float w2s  = w2 * sh0 * INV_AVG;

        unsafeAtomicAdd(&aggs[(size_t)r * 64 + u],       ms0);
        unsafeAtomicAdd(&aggs[(size_t)r * 64 + 32 + u],  ms1);
        unsafeAtomicAdd(&aggv[((size_t)r * 3 + 0) * 64 + u],      w1x * s1x);
        unsafeAtomicAdd(&aggv[((size_t)r * 3 + 1) * 64 + u],      w1x * s1y);
        unsafeAtomicAdd(&aggv[((size_t)r * 3 + 2) * 64 + u],      w1x * s1z);
        unsafeAtomicAdd(&aggv[((size_t)r * 3 + 0) * 64 + 32 + u], w2s * xvx);
        unsafeAtomicAdd(&aggv[((size_t)r * 3 + 1) * 64 + 32 + u], w2s * xvy);
        unsafeAtomicAdd(&aggv[((size_t)r * 3 + 2) * 64 + 32 + u], w2s * xvz);
    }
}

// ---------------------------------------------------------------------------
// Kernel C: node post-linears + self-connection + gating. One node per block.
// t<64: out_s channel t;  64<=t<160: out_v (i=(t-64)>>5, v=(t-64)&31)
// ---------------------------------------------------------------------------
__global__ __launch_bounds__(160)
void node_post(const float* __restrict__ nf, const float* __restrict__ attrs,
               const float* __restrict__ aggs, const float* __restrict__ aggv,
               const float* __restrict__ W2s, const float* __restrict__ W2v,
               const float* __restrict__ Wsc_s, const float* __restrict__ Wsc_v,
               float* __restrict__ out) {
    __shared__ float s_attr[4], s_nf[128], s_aggs[64], s_aggv[192], s_outs[64];
    const int n = blockIdx.x, t = threadIdx.x;
    for (int i = t; i < 128; i += 160) s_nf[i]   = nf[(size_t)n * 128 + i];
    for (int i = t; i < 64;  i += 160) s_aggs[i] = aggs[(size_t)n * 64 + i];
    for (int i = t; i < 192; i += 160) s_aggv[i] = aggv[(size_t)n * 192 + i];
    if (t < 4) s_attr[t] = attrs[(size_t)n * 4 + t];
    __syncthreads();

    float outv = 0.f;
    int i3 = 0, v = 0;
    if (t < 64) {
        float f = 0.f;
#pragma unroll 4
        for (int u = 0; u < 64; ++u) f += s_aggs[u] * W2s[u * 64 + t];
        f *= INV_8;
#pragma unroll 2
        for (int u = 0; u < 32; ++u) {
            const float xu = s_nf[u] * NORM_SC;
#pragma unroll
            for (int a = 0; a < 4; ++a)
                f += xu * s_attr[a] * Wsc_s[u * 256 + a * 64 + t];
        }
        s_outs[t] = f;
    } else {
        i3 = (t - 64) >> 5;
        v  = (t - 64) & 31;
        float g = 0.f;
#pragma unroll 4
        for (int u = 0; u < 64; ++u) g += s_aggv[i3 * 64 + u] * W2v[u * 32 + v];
        g *= INV_8;
#pragma unroll 2
        for (int u = 0; u < 32; ++u) {
            const float xui = s_nf[32 + u * 3 + i3] * NORM_SC;
#pragma unroll
            for (int a = 0; a < 4; ++a)
                g += xui * s_attr[a] * Wsc_v[u * 128 + a * 32 + v];
        }
        outv = g;
    }
    __syncthreads();

    if (t < 32) {
        out[(size_t)n * 128 + t] = s_nf[t] + silu_f(s_outs[t]);
    } else if (t >= 64) {
        const float gate = silu_f(s_outs[32 + v]);
        const int idx = 32 + v * 3 + i3;
        out[(size_t)n * 128 + idx] = s_nf[idx] + outv * gate;
    }
}

// ---------------------------------------------------------------------------
// Launch
// ---------------------------------------------------------------------------
extern "C" void kernel_launch(void* const* d_in, const int* in_sizes, int n_in,
                              void* d_out, int out_size, void* d_ws, size_t ws_size,
                              hipStream_t stream) {
    const float* nf     = (const float*)d_in[0];
    const float* attrs  = (const float*)d_in[1];
    const float* ef     = (const float*)d_in[2];
    const float* ea     = (const float*)d_in[3];
    const float* W1s    = (const float*)d_in[4];
    const float* W1v    = (const float*)d_in[5];
    const float* Wf1    = (const float*)d_in[6];
    const float* Wf2    = (const float*)d_in[7];
    const float* Wf3    = (const float*)d_in[8];
    const float* W2s    = (const float*)d_in[9];
    const float* W2v    = (const float*)d_in[10];
    const float* Wsc_s  = (const float*)d_in[11];
    const float* Wsc_v  = (const float*)d_in[12];
    const int*   eidx   = (const int*)d_in[13];

    const int N = in_sizes[0] / 128;   // 32768
    const int E = in_sizes[2] / N_RAD; // 524288

    float* ws = (float*)d_ws;
    float* xs   = ws;                           // N*32
    float* xv   = ws + (size_t)N * 32;          // N*96
    float* aggs = ws + (size_t)N * 128;         // N*64
    float* aggv = ws + (size_t)N * 192;         // N*192
    _Float16* wall = (_Float16*)(ws + (size_t)N * 384);  // 14336 f16

    prep_weights<<<1, 256, 0, stream>>>(Wf1, Wf2, Wf3, wall);
    node_pre<<<N / 8, 256, 0, stream>>>(nf, W1s, W1v, xs, xv);
    (void)hipMemsetAsync(aggs, 0, (size_t)N * 256 * sizeof(float), stream);
    edge_kernel<<<E / (WPB * 16), WPB * 32, 0, stream>>>(
        ef, ea, eidx, xs, xv, wall, aggs, aggv, E);
    node_post<<<N, 160, 0, stream>>>(nf, attrs, aggs, aggv, W2s, W2v,
                                     Wsc_s, Wsc_v, (float*)d_out);
}
